// EMA_VFI_52553219834669
// MI455X (gfx1250) — matrix operand order repaired and checked by reading, not compiled, not run on hardware
//
#include <hip/hip_runtime.h>
#include <hip/hip_bf16.h>

typedef __attribute__((ext_vector_type(16))) _Float16 v16h;
typedef __attribute__((ext_vector_type(8)))  float    v8f;

// ---- WMMA fragment index helpers (wave32, contiguous k order chosen by this code) -----------------
// A (16x32 f16), this code's k order: lanes 0-15 -> M=lane, K=i; lanes 16-31 -> M=lane-16, K=i+16
__device__ __forceinline__ int khalfA(int i, int lane) {
  return i + (lane & 16);
}
// B (32x16 f16), this code's k order: lanes 0-15 -> N=lane, K=i; lanes 16-31 -> N=lane-16, K=i+16
__device__ __forceinline__ int khalfB(int i, int lane) {
  return i + (lane & 16);
}
__device__ __forceinline__ int iclamp(int v, int lo, int hi) {
  return v < lo ? lo : (v > hi ? hi : v);
}

// ---------------------------------------------------------------------------
// Weight packer: OIHW f32 -> [ocTile][cinChunk][tap(9)][lane(32)][i(16)] f16.
// Taps are innermost-contiguous: inside the conv loop the 9 A fragments for a
// given (ocTile, chunk) live at base + k*1024B -> literal-immediate loads.
// ---------------------------------------------------------------------------
__global__ void pack_w_kernel(const float* __restrict__ w, _Float16* __restrict__ wp,
                              int Cin, int Cout)
{
  const int octiles = (Cout + 15) >> 4;
  const int chunks  = (Cin + 31) >> 5;
  const long total  = 9L * octiles * chunks * 512;
  const long idx = (long)blockIdx.x * blockDim.x + threadIdx.x;
  if (idx >= total) return;
  const int i    = (int)(idx & 15);
  const int lane = (int)((idx >> 4) & 31);
  long r = idx >> 9;
  const int k  = (int)(r % 9);      r /= 9;
  const int ch = (int)(r % chunks); r /= chunks;
  const int ot = (int)r;
  const int oc  = (ot << 4) + (lane & 15);
  const int cin = (ch << 5) + khalfA(i, lane);
  float v = 0.0f;
  if (oc < Cout && cin < Cin) v = w[((long)oc * Cin + cin) * 9 + k];
  wp[idx] = (_Float16)v;
}

// ---------------------------------------------------------------------------
// Implicit-GEMM 3x3 conv, pad=1, stride in {1,2}.
// Block = 256 threads = 8 waves: wave -> (ocTile = wave>>1, pxPair = wave&1).
// Each wave owns TWO 16x16 output tiles (2 accumulators) sharing one A
// fragment per tap -> 2 WMMAs per A load. Block tile = 64 oc x 64 px.
// Activations staged per-cin-chunk in LDS as [row][px][cin(+pad)] f16 so each
// B fragment is one contiguous 32B LDS read. A fragments come pre-packed with
// taps contiguous -> base pointer + compile-time immediate offsets.
// act: 0=none, 1=relu, 2=(tanh(x)+1)/2
// ---------------------------------------------------------------------------
#define SH_PW 130  // 64*stride + 2, stride<=2
#define SH_STR 40  // padded cin stride (halves): 16B-aligned, conflict-free b128
__global__ __launch_bounds__(256) void conv3x3_wmma_kernel(
    const float* __restrict__ in, const _Float16* __restrict__ wp,
    const float* __restrict__ bias, float* __restrict__ out,
    int B, int Cin, int Cout, int H, int W, int stride, int act)
{
  __shared__ __align__(32) _Float16 sh[3 * SH_PW * SH_STR];

  const int Ho = (H - 1) / stride + 1;
  const int Wo = (W - 1) / stride + 1;
  const int octiles = (Cout + 15) >> 4;
  const int chunks  = (Cin + 31) >> 5;
  const int ogroups = (octiles + 3) >> 2;
  const int xgroups = (Wo + 63) >> 6;
  const int PW = 64 * stride + 2;

  int t = blockIdx.x;
  const int og = t % ogroups; t /= ogroups;
  const int xg = t % xgroups; t /= xgroups;
  const int y  = t % Ho;      t /= Ho;
  const int b  = t;

  const int tid  = threadIdx.x;
  const int lane = tid & 31;
  const int wave = tid >> 5;
  const int pth  = wave & 1;            // pixel-tile pair within block
  const int oct  = (og << 2) + (wave >> 1);
  const bool active = oct < octiles;

  const int mn  = lane & 15;
  const int kh  = lane & 16;
  const int pt0 = (pth << 1);           // first 16-px tile of this wave
  const int xo0 = (xg << 6) + (pt0 << 4) + mn;        // output x, tile 0
  const int xo1 = xo0 + 16;                           // output x, tile 1
  const int x0s = (xg << 6) * stride;                 // input x of strip origin (+1 pad)

  // per-wave packed-weight base for chunk 0 (taps contiguous at +k*512 halves)
  const _Float16* wpa0 = wp + ((long)oct * chunks * 9) * 512 + lane * 16;

  v8f acc0 = {}, acc1 = {};
  for (int ch = 0; ch < chunks; ++ch) {
    __syncthreads();
    // ---- cooperative stage: 32 cin x 3 rows x PW px (clamped, masked) ----
    const int nElem = 96 * PW;
    for (int idx = tid; idx < nElem; idx += 256) {
      const int c  = idx / (3 * PW);
      const int rp = idx % (3 * PW);
      const int r  = rp / PW;
      const int px = rp % PW;
      const int cin = (ch << 5) + c;
      const int iy = y * stride + r - 1;
      const int ix = x0s + px - 1;
      const bool ok = (iy >= 0) & (iy < H) & (ix >= 0) & (ix < W) & (cin < Cin);
      const long ga = (((long)b * Cin + (cin < Cin ? cin : Cin - 1)) * H
                       + iclamp(iy, 0, H - 1)) * W + iclamp(ix, 0, W - 1);
      const float v = in[ga];                       // unconditional clamped load
      sh[(r * PW + px) * SH_STR + c] = (_Float16)(ok ? v : 0.0f);
    }
    __syncthreads();
    // ---- 9 taps: one A fragment feeds two WMMAs (2 pixel tiles) ----
    if (active) {
      const _Float16* wpa = wpa0 + (long)ch * 9 * 512;
#pragma unroll
      for (int k = 0; k < 9; ++k) {
        const v16h af = *(const v16h*)(wpa + k * 512);   // imm-offset b128 pair
        const int r   = k / 3;
        const int px0 = (pt0 * 16 + mn) * stride + (k % 3);
        const v16h bf0 = *(const v16h*)(&sh[(r * PW + px0) * SH_STR + kh]);
        const v16h bf1 = *(const v16h*)(&sh[(r * PW + px0 + 16 * stride) * SH_STR + kh]);
        acc0 = __builtin_amdgcn_wmma_f32_16x16x32_f16(
            false, af, false, bf0, (short)0, acc0, false, false);
        acc1 = __builtin_amdgcn_wmma_f32_16x16x32_f16(
            false, af, false, bf1, (short)0, acc1, false, false);
      }
    }
  }

  if (active) {
#pragma unroll
    for (int rr = 0; rr < 8; ++rr) {
      const int oc = (oct << 4) + rr + (kh >> 1);
      if (oc < Cout) {
        const long rowBase = (((long)b * Cout + oc) * Ho + y) * Wo;
        float v0 = acc0[rr] + bias[oc];
        float v1 = acc1[rr] + bias[oc];
        if (act == 1)      { v0 = fmaxf(v0, 0.0f); v1 = fmaxf(v1, 0.0f); }
        else if (act == 2) { v0 = (tanhf(v0) + 1.0f) * 0.5f; v1 = (tanhf(v1) + 1.0f) * 0.5f; }
        if (xo0 < Wo) out[rowBase + xo0] = v0;
        if (xo1 < Wo) out[rowBase + xo1] = v1;
      }
    }
  }
}

// ---------------------------------------------------------------------------
// Modulated deformable conv (torchvision v2), k=3,s=1,p=1, C->C.
// Bilinear gathers are unconditional (clamped corner indices); validity and the
// sigmoid mask are folded into the 4 corner weights -> no exec-mask branches.
// ---------------------------------------------------------------------------
__global__ __launch_bounds__(256) void dcn_wmma_kernel(
    const float* __restrict__ in, const float* __restrict__ off27,
    const _Float16* __restrict__ wp, const float* __restrict__ bias,
    float* __restrict__ out, int B, int C, int H, int W)
{
  const int tiles_x = (W + 15) >> 4;
  const int octiles = (C + 15) >> 4;
  const int chunks  = (C + 31) >> 5;
  const int lane = threadIdx.x & 31;

  long tile = (long)blockIdx.x * (blockDim.x >> 5) + (threadIdx.x >> 5);
  int oct = (int)(tile % octiles); tile /= octiles;
  int tx  = (int)(tile % tiles_x); tile /= tiles_x;
  int y   = (int)(tile % H);       tile /= H;
  int b   = (int)tile;
  if (b >= B) return;                       // uniform per wave

  const int mn = lane & 15;
  const int kh = lane & 16;
  const int xo = (tx << 4) + mn;
  const int xc = xo < W ? xo : W - 1;
  const long hw = (long)H * W;
  const float* offb = off27 + (long)b * 27 * hw;
  const float* inb  = in + (long)b * C * hw;
  const _Float16* wpa0 = wp + ((long)oct * chunks * 9) * 512 + lane * 16;

  v8f acc = {};
  for (int k = 0; k < 9; ++k) {
    int cdy = 2 * k;     cdy = (cdy < 9) ? cdy : cdy + 9;
    int cdx = 2 * k + 1; cdx = (cdx < 9) ? cdx : cdx + 9;
    const long pidx = (long)y * W + xc;
    const float dy = offb[(long)cdy * hw + pidx];
    const float dx = offb[(long)cdx * hw + pidx];
    const float mk = 1.0f / (1.0f + expf(-offb[(long)(9 + k) * hw + pidx]));
    const float py = (float)y + (float)(k / 3 - 1) + dy;
    const float px = (float)xc + (float)(k % 3 - 1) + dx;

    const float y0f = floorf(py), x0f = floorf(px);
    const int y0 = (int)y0f, x0 = (int)x0f;
    const float wy = py - y0f, wx = px - x0f;
    const bool oky0 = (y0 >= 0) && (y0 < H);
    const bool oky1 = (y0 + 1 >= 0) && (y0 + 1 < H);
    const bool okx0 = (x0 >= 0) && (x0 < W);
    const bool okx1 = (x0 + 1 >= 0) && (x0 + 1 < W);
    const float m00 = (oky0 && okx0) ? (1.0f - wy) * (1.0f - wx) * mk : 0.0f;
    const float m01 = (oky0 && okx1) ? (1.0f - wy) * wx * mk : 0.0f;
    const float m10 = (oky1 && okx0) ? wy * (1.0f - wx) * mk : 0.0f;
    const float m11 = (oky1 && okx1) ? wy * wx * mk : 0.0f;
    const long yi0 = (long)iclamp(y0,     0, H - 1) * W;
    const long yi1 = (long)iclamp(y0 + 1, 0, H - 1) * W;
    const long xi0 = iclamp(x0,     0, W - 1);
    const long xi1 = iclamp(x0 + 1, 0, W - 1);
    const long i00 = yi0 + xi0, i01 = yi0 + xi1, i10 = yi1 + xi0, i11 = yi1 + xi1;

    for (int ch = 0; ch < chunks; ++ch) {
      const v16h af = *(const v16h*)(wpa0 + ((long)ch * 9 + k) * 512);
      v16h bf;
#pragma unroll
      for (int i = 0; i < 16; ++i) {
        const int cinB = (ch << 5) + i + kh;     // khalfB is contiguous in i
        const float* pl = inb + (long)(cinB < C ? cinB : C - 1) * hw;
        const float s = m00 * pl[i00] + m01 * pl[i01] + m10 * pl[i10] + m11 * pl[i11];
        bf[i] = (_Float16)(cinB < C ? s : 0.0f);
      }
      acc = __builtin_amdgcn_wmma_f32_16x16x32_f16(
          false, af, false, bf, (short)0, acc, false, false);
    }
  }

#pragma unroll
  for (int r = 0; r < 8; ++r) {
    const int oc = (oct << 4) + r + (kh >> 1);
    if (oc < C && xo < W)
      out[(((long)b * C + oc) * H + y) * W + xo] = acc[r] + bias[oc];
  }
}

// ---------------------------------------------------------------------------
// Glue kernels (bandwidth-trivial)
// ---------------------------------------------------------------------------
__global__ void concat2_kernel(const float* __restrict__ a, const float* __restrict__ b_,
                               float* __restrict__ out, int B, int Ca, int Cb, long hw)
{
  const long idx = (long)blockIdx.x * blockDim.x + threadIdx.x;
  const long total = (long)B * (Ca + Cb) * hw;
  if (idx >= total) return;
  const long p = idx % hw;
  long t = idx / hw;
  const int c = (int)(t % (Ca + Cb));
  const int b = (int)(t / (Ca + Cb));
  out[idx] = (c < Ca) ? a[((long)b * Ca + c) * hw + p]
                      : b_[((long)b * Cb + (c - Ca)) * hw + p];
}

__global__ void concat_ctx_kernel(const float* __restrict__ feat, const float* __restrict__ ctx,
                                  float* __restrict__ out, int B, long hw)
{
  const long idx = (long)blockIdx.x * blockDim.x + threadIdx.x;
  const long total = (long)B * 128 * hw;
  if (idx >= total) return;
  const long p = idx % hw;
  long t = idx / hw;
  const int c = (int)(t % 128);
  const int b = (int)(t / 128);
  out[idx] = (c < 64) ? feat[((long)b * 64 + c) * hw + p] : ctx[b * 64 + (c - 64)];
}

__global__ void warp_kernel(const float* __restrict__ frame2, const float* __restrict__ flow,
                            float* __restrict__ out, int B, int H, int W)
{
  const long hw = (long)H * W;
  const long idx = (long)blockIdx.x * blockDim.x + threadIdx.x;
  const long total = (long)B * 3 * hw;
  if (idx >= total) return;
  const int x = (int)(idx % W);
  long t = idx / W;
  const int y = (int)(t % H); t /= H;
  const int c = (int)(t % 3);
  const int b = (int)(t / 3);
  const float px = (float)x + flow[((long)b * 2 + 0) * hw + (long)y * W + x];
  const float py = (float)y + flow[((long)b * 2 + 1) * hw + (long)y * W + x];
  const float y0f = floorf(py), x0f = floorf(px);
  const int y0 = (int)y0f, x0 = (int)x0f;
  const float wy = py - y0f, wx = px - x0f;
  const bool oky0 = (y0 >= 0) && (y0 < H), oky1 = (y0 + 1 >= 0) && (y0 + 1 < H);
  const bool okx0 = (x0 >= 0) && (x0 < W), okx1 = (x0 + 1 >= 0) && (x0 + 1 < W);
  const float m00 = (oky0 && okx0) ? (1 - wy) * (1 - wx) : 0.0f;
  const float m01 = (oky0 && okx1) ? (1 - wy) * wx : 0.0f;
  const float m10 = (oky1 && okx0) ? wy * (1 - wx) : 0.0f;
  const float m11 = (oky1 && okx1) ? wy * wx : 0.0f;
  const long yi0 = (long)iclamp(y0, 0, H - 1) * W, yi1 = (long)iclamp(y0 + 1, 0, H - 1) * W;
  const long xi0 = iclamp(x0, 0, W - 1), xi1 = iclamp(x0 + 1, 0, W - 1);
  const float* img = frame2 + ((long)b * 3 + c) * hw;
  out[idx] = m00 * img[yi0 + xi0] + m01 * img[yi0 + xi1] +
             m10 * img[yi1 + xi0] + m11 * img[yi1 + xi1];
}

__global__ void avgpool_kernel(const float* __restrict__ in, float* __restrict__ out, int HW)
{
  __shared__ float sd[256];
  const int bc = blockIdx.x;
  float s = 0.0f;
  for (int i = threadIdx.x; i < HW; i += blockDim.x) s += in[(long)bc * HW + i];
  sd[threadIdx.x] = s;
  __syncthreads();
  for (int st = 128; st > 0; st >>= 1) {
    if ((int)threadIdx.x < st) sd[threadIdx.x] += sd[threadIdx.x + st];
    __syncthreads();
  }
  if (threadIdx.x == 0) out[bc] = sd[0] / (float)HW;
}

__global__ void linear_kernel(const float* __restrict__ pooled, const float* __restrict__ lw,
                              const float* __restrict__ lb, float* __restrict__ ctx,
                              int B, int In, int Out)
{
  const int t = blockIdx.x * blockDim.x + threadIdx.x;
  if (t >= B * Out) return;
  const int b = t / Out, o = t % Out;
  float s = lb[o];
  for (int i = 0; i < In; ++i) s += pooled[b * In + i] * lw[o * In + i];
  ctx[t] = s;
}

// ---------------------------------------------------------------------------
// Host-side pipeline
// ---------------------------------------------------------------------------
static inline long pack_halves(int Cin, int Cout) {
  return 9L * ((Cout + 15) / 16) * ((Cin + 31) / 32) * 512;
}
static inline _Float16* take_pack(float*& cur, int Cin, int Cout) {
  _Float16* p = (_Float16*)cur;
  cur += pack_halves(Cin, Cout) / 2;   // halves are multiples of 512 -> even
  return p;
}
static inline void launch_pack(const float* w, _Float16* wp, int Cin, int Cout, hipStream_t s) {
  const long total = pack_halves(Cin, Cout);
  pack_w_kernel<<<(unsigned)((total + 255) / 256), 256, 0, s>>>(w, wp, Cin, Cout);
}
static inline void launch_conv(const float* in, const _Float16* wp, const float* b, float* out,
                               int B, int Cin, int Cout, int H, int W, int stride, int act,
                               hipStream_t s)
{
  const int Ho = (H - 1) / stride + 1;
  const int Wo = (W - 1) / stride + 1;
  const long blocks = (long)B * Ho * ((Wo + 63) / 64) * ((((Cout + 15) / 16) + 3) / 4);
  conv3x3_wmma_kernel<<<(unsigned)blocks, 256, 0, s>>>(in, wp, b, out, B, Cin, Cout, H, W, stride, act);
}
static inline void launch_dcn(const float* in, const float* off27, const _Float16* wp,
                              const float* b, float* out, int B, int C, int H, int W,
                              hipStream_t s)
{
  const long tiles = (long)B * H * ((W + 15) / 16) * ((C + 15) / 16);
  dcn_wmma_kernel<<<(unsigned)((tiles + 7) / 8), 256, 0, s>>>(in, off27, wp, b, out, B, C, H, W);
}
static inline unsigned ew_blocks(long n) { return (unsigned)((n + 255) / 256); }

extern "C" void kernel_launch(void* const* d_in, const int* in_sizes, int n_in,
                              void* d_out, int out_size, void* d_ws, size_t ws_size,
                              hipStream_t stream)
{
  (void)in_sizes; (void)n_in; (void)out_size; (void)ws_size;
  const int B = 2, H = 224, W = 224;
  const long S = (long)H * W;

  const float* frame1 = (const float*)d_in[0];
  const float* frame2 = (const float*)d_in[1];
  const float* fe_w[4] = {(const float*)d_in[2], (const float*)d_in[3], (const float*)d_in[4], (const float*)d_in[5]};
  const float* fe_b[4] = {(const float*)d_in[6], (const float*)d_in[7], (const float*)d_in[8], (const float*)d_in[9]};
  const float* ce_w[3] = {(const float*)d_in[10], (const float*)d_in[11], (const float*)d_in[12]};
  const float* ce_b[3] = {(const float*)d_in[13], (const float*)d_in[14], (const float*)d_in[15]};
  const float* ce_lw = (const float*)d_in[16];
  const float* ce_lb = (const float*)d_in[17];
  const float* me_w[3] = {(const float*)d_in[18], (const float*)d_in[19], (const float*)d_in[20]};
  const float* me_b[3] = {(const float*)d_in[21], (const float*)d_in[22], (const float*)d_in[23]};
  const float* ao_w[3] = {(const float*)d_in[24], (const float*)d_in[25], (const float*)d_in[26]};
  const float* ao_b[3] = {(const float*)d_in[27], (const float*)d_in[28], (const float*)d_in[29]};
  const float* ad_w[3] = {(const float*)d_in[30], (const float*)d_in[31], (const float*)d_in[32]};
  const float* ad_b[3] = {(const float*)d_in[33], (const float*)d_in[34], (const float*)d_in[35]};
  const float* rc_w[3] = {(const float*)d_in[36], (const float*)d_in[37], (const float*)d_in[38]};
  const float* rc_b[3] = {(const float*)d_in[39], (const float*)d_in[40], (const float*)d_in[41]};

  float* ws = (float*)d_ws;
  float* P    = ws;                       // 2*67*S  ping buffer
  float* F    = P  + (long)2 * 67 * S;    // 2*67*S  pong buffer (feat)
  float* G    = F  + (long)2 * 67 * S;    // 2*128*S big buffer (fi / fused)
  float* C1   = G  + (long)2 * 128 * S;   // 2*128*112*112 (ce1 / ce3 / off27)
  float* C2   = C1 + (long)2 * 128 * 112 * 112; // 2*256*56*56
  float* IN6  = C2 + (long)2 * 256 * 56 * 56;   // 2*6*S
  float* POOL = IN6 + (long)2 * 6 * S;    // 512
  float* CTX  = POOL + 512;               // 128
  float* FLOW = CTX + 128;                // 2*2*S
  float* WARP = FLOW + (long)2 * 2 * S;   // 2*3*S
  float* cur  = WARP + (long)2 * 3 * S;   // packed-weight arena (~3.2 MB)
  float* OUT  = (float*)d_out;

  // packed weight buffers
  _Float16* p_fe[4]; for (int i = 0; i < 4; ++i) p_fe[i] = take_pack(cur, i ? 64 : 6, 64);
  _Float16* p_ce0 = take_pack(cur, 64, 128);
  _Float16* p_ce1 = take_pack(cur, 128, 256);
  _Float16* p_ce2 = take_pack(cur, 256, 256);
  _Float16* p_me0 = take_pack(cur, 128, 64);
  _Float16* p_me1 = take_pack(cur, 64, 64);
  _Float16* p_me2 = take_pack(cur, 64, 2);
  _Float16* p_ao[3]; for (int i = 0; i < 3; ++i) p_ao[i] = take_pack(cur, 67, 27);
  _Float16* p_ad[3]; for (int i = 0; i < 3; ++i) p_ad[i] = take_pack(cur, 67, 67);
  _Float16* p_rc0 = take_pack(cur, 67, 64);
  _Float16* p_rc1 = take_pack(cur, 64, 32);
  _Float16* p_rc2 = take_pack(cur, 32, 3);

  // pack all weights (cheap; runs every call -> deterministic)
  for (int i = 0; i < 4; ++i) launch_pack(fe_w[i], p_fe[i], i ? 64 : 6, 64, stream);
  launch_pack(ce_w[0], p_ce0, 64, 128, stream);
  launch_pack(ce_w[1], p_ce1, 128, 256, stream);
  launch_pack(ce_w[2], p_ce2, 256, 256, stream);
  launch_pack(me_w[0], p_me0, 128, 64, stream);
  launch_pack(me_w[1], p_me1, 64, 64, stream);
  launch_pack(me_w[2], p_me2, 64, 2, stream);
  for (int i = 0; i < 3; ++i) launch_pack(ao_w[i], p_ao[i], 67, 27, stream);
  for (int i = 0; i < 3; ++i) launch_pack(ad_w[i], p_ad[i], 67, 67, stream);
  launch_pack(rc_w[0], p_rc0, 67, 64, stream);
  launch_pack(rc_w[1], p_rc1, 64, 32, stream);
  launch_pack(rc_w[2], p_rc2, 32, 3, stream);

  // 1. input concat [B,6,H,W]
  concat2_kernel<<<ew_blocks((long)B * 6 * S), 256, 0, stream>>>(frame1, frame2, IN6, B, 3, 3, S);

  // 2. feature extractor -> feat in F
  launch_conv(IN6, p_fe[0], fe_b[0], P, B, 6, 64, H, W, 1, 1, stream);
  launch_conv(P,   p_fe[1], fe_b[1], F, B, 64, 64, H, W, 1, 1, stream);
  launch_conv(F,   p_fe[2], fe_b[2], P, B, 64, 64, H, W, 1, 1, stream);
  launch_conv(P,   p_fe[3], fe_b[3], F, B, 64, 64, H, W, 1, 1, stream);

  // 3. context encoder + pooled linear
  launch_conv(F,  p_ce0, ce_b[0], C1, B, 64, 128, H, W, 2, 1, stream);      // 112x112
  launch_conv(C1, p_ce1, ce_b[1], C2, B, 128, 256, 112, 112, 2, 1, stream); // 56x56
  launch_conv(C2, p_ce2, ce_b[2], C1, B, 256, 256, 56, 56, 1, 1, stream);
  avgpool_kernel<<<B * 256, 256, 0, stream>>>(C1, POOL, 56 * 56);
  linear_kernel<<<1, 128, 0, stream>>>(POOL, ce_lw, ce_lb, CTX, B, 256, 64);

  // 4. motion encoder -> flow
  concat_ctx_kernel<<<ew_blocks((long)B * 128 * S), 256, 0, stream>>>(F, CTX, G, B, S);
  launch_conv(G, p_me0, me_b[0], P, B, 128, 64, H, W, 1, 1, stream);
  launch_conv(P, p_me1, me_b[1], G, B, 64, 64, H, W, 1, 1, stream);
  launch_conv(G, p_me2, me_b[2], FLOW, B, 64, 2, H, W, 1, 0, stream);

  // 5. warp frame2 with flow
  warp_kernel<<<ew_blocks((long)B * 3 * S), 256, 0, stream>>>(frame2, FLOW, WARP, B, H, W);

  // 6. fused = [feat, warped] -> 3x MDCP
  concat2_kernel<<<ew_blocks((long)B * 67 * S), 256, 0, stream>>>(F, WARP, G, B, 64, 3, S);
  launch_conv(G, p_ao[0], ao_b[0], C1, B, 67, 27, H, W, 1, 0, stream);
  launch_dcn(G, C1, p_ad[0], ad_b[0], P, B, 67, H, W, stream);
  launch_conv(P, p_ao[1], ao_b[1], C1, B, 67, 27, H, W, 1, 0, stream);
  launch_dcn(P, C1, p_ad[1], ad_b[1], G, B, 67, H, W, stream);
  launch_conv(G, p_ao[2], ao_b[2], C1, B, 67, 27, H, W, 1, 0, stream);
  launch_dcn(G, C1, p_ad[2], ad_b[2], P, B, 67, H, W, stream);

  // 7. reconstruction -> d_out
  launch_conv(P, p_rc0, rc_b[0], G, B, 67, 64, H, W, 1, 1, stream);
  launch_conv(G, p_rc1, rc_b[1], F, B, 64, 32, H, W, 1, 1, stream);
  launch_conv(F, p_rc2, rc_b[2], OUT, B, 32, 3, H, W, 1, 2, stream);
}
